// EagerLSTM_3728031613797
// MI455X (gfx1250) — matrix operand order repaired and checked by reading, not compiled, not run on hardware
//
#include <hip/hip_runtime.h>
#include <hip/hip_bf16.h>

// ---------------------------------------------------------------------------
// LSTM forward, B=32 T=512 D=512 H=1024, gate order i,f,g,o, with the quirk
// c_new = tanh(f*c + i*g) applied before carry.
//
// Strategy (MI455X / gfx1250, wave32, WMMA):
//  - Prep kernel: feats -> bf16; fragment-major bf16 copy of [Wx; Wh]
//    (K=1536, N=4096) in contiguous-k B-operand order (k = 16*(lane/16) + elem), same as A.
//  - Persistent kernel: 64 blocks x 256 threads = 512 waves, one 16x16 gate
//    tile per wave. Weights: each wave holds its 48 B fragments in 384 VGPRs
//    (loaded once; wave32 supports 1024 VGPRs) -> zero weight traffic in the
//    recurrence. Activations: all 8 waves of a block share one M-tile, so
//    per step the block async-copies its 16 x-rows + 16 h-rows (48 KB) into
//    LDS via global_load_async_to_lds_b128 (ASYNCcnt), then the 48-WMMA loop
//    reads A fragments with ds_load (padded rows to spread banks).
//    The LDS destination operand is produced by ptrtoint of the real smem
//    pointer so the buffer escapes and the compiler keeps/orders the LDS
//    reads (a raw hand-computed offset let LLVM fold them to undef).
//    Per timestep: stage -> WMMA phase -> grid barrier -> update -> barrier.
// ---------------------------------------------------------------------------

typedef __attribute__((ext_vector_type(16))) __bf16 v16bf;
typedef __attribute__((ext_vector_type(8)))  __bf16 v8bf;
typedef __attribute__((ext_vector_type(8)))  float  v8f;

#define B_   32
#define T_   512
#define D_   512
#define H_   1024
#define G_   4096            // 4*H
#define K_   1536            // D + H
#define KB_  48              // K_/32 k-blocks per tile
#define KBX_ 16              // x-part k-blocks (D/32)
#define KBH_ 32              // h-part k-blocks (H/32)
#define NT_  256             // G_/16 N-tiles
#define NBLOCKS_  64
#define NTHREADS_ 256

// LDS staging (bf16 elems); +8 pad per row => row stride shifts 4 banks
#define XPITCH 520           // 512 + 8
#define HPITCH 1032          // 1024 + 8
#define SMEM_X_ELEMS (16 * XPITCH)
#define SMEM_H_ELEMS (16 * HPITCH)

// workspace layout (bytes); all offsets multiple of 256
#define OFF_BAR     0
#define OFF_WFRAG   256
#define WFRAG_ELEMS ((size_t)KB_ * NT_ * 512)            // bf16 elems (12.6 MB)
#define OFF_FEATSB  (OFF_WFRAG + WFRAG_ELEMS * 2)
#define FEATSB_ELEMS ((size_t)B_ * T_ * D_)              // bf16 elems (16.8 MB)
#define OFF_HB      (OFF_FEATSB + FEATSB_ELEMS * 2)
#define OFF_C       (OFF_HB + (size_t)B_ * H_ * 2)
#define OFF_GATES   (OFF_C + (size_t)B_ * H_ * 4)

__device__ __forceinline__ unsigned short f2bfbits(float x) {
  union { float f; unsigned u; } v; v.f = x;
  // round-to-nearest-even f32 -> bf16
  return (unsigned short)((v.u + 0x7FFFu + ((v.u >> 16) & 1u)) >> 16);
}

// -------------------------- prep kernel ------------------------------------
__global__ void lstm_prep(const float* __restrict__ feats,
                          const float* __restrict__ Wx,
                          const float* __restrict__ Wh,
                          unsigned char* __restrict__ ws) {
  unsigned gid = blockIdx.x * blockDim.x + threadIdx.x;
  unsigned short* wfrag  = (unsigned short*)(ws + OFF_WFRAG);
  unsigned short* featsb = (unsigned short*)(ws + OFF_FEATSB);
  unsigned short* hb     = (unsigned short*)(ws + OFF_HB);
  float*          cst    = (float*)(ws + OFF_C);

  // Weight -> WMMA B-fragment layout. B is 32x16 (KxN) per tile:
  // lanes 0-15 hold K=0..15 (2 per VGPR lo/hi), lanes 16-31 hold K=16..31,
  // column n = lane&15. Tile = 32 lanes * 16 bf16, contiguous per lane.
  if (gid < (unsigned)(K_ * G_)) {
    int k = (int)(gid / G_);
    int n = (int)(gid % G_);
    float v = (k < D_) ? Wx[(size_t)k * G_ + n]
                       : Wh[(size_t)(k - D_) * G_ + n];
    int kb = k >> 5, kk = k & 31, nt = n >> 4, nn = n & 15;
    int lane = nn + ((kk >> 4) << 4);
    int vreg = (kk & 15) >> 1, half = kk & 1;
    wfrag[(size_t)(kb * NT_ + nt) * 512 + lane * 16 + vreg * 2 + half] =
        f2bfbits(v);
  }
  if (gid < (unsigned)FEATSB_ELEMS) featsb[gid] = f2bfbits(feats[gid]);
  if (gid < (unsigned)(B_ * H_)) { hb[gid] = 0; cst[gid] = 0.0f; }
  if (gid == 0) {
    ((unsigned*)(ws + OFF_BAR))[0] = 0;  // arrival counter
    ((unsigned*)(ws + OFF_BAR))[1] = 0;  // generation
  }
}

// ------------------------ grid-wide barrier --------------------------------
__device__ __forceinline__ void gridBarrier(unsigned* bar) {
  __syncthreads();
  __threadfence();
  if (threadIdx.x == 0) {
    volatile unsigned* vgen = bar + 1;
    unsigned g = *vgen;
    if (atomicAdd(bar, 1u) == NBLOCKS_ - 1u) {
      bar[0] = 0;
      __threadfence();
      *vgen = g + 1u;
    } else {
      while (*vgen == g) { __builtin_amdgcn_s_sleep(1); }
    }
  }
  __threadfence();
  __syncthreads();
}

// async 16B global -> LDS copy (CDNA5; tracked by ASYNCcnt).
// lds_addr comes from ptrtoint of the real LDS pointer (low 32 bits of the
// flat shared address are the wave-relative LDS byte address) so the shared
// buffer is captured and LLVM keeps the subsequent LDS reads.
__device__ __forceinline__ void asyncCopy16(unsigned lds_addr,
                                            const void* gsrc) {
  unsigned long long ga = (unsigned long long)gsrc;
  asm volatile("global_load_async_to_lds_b128 %0, %1, off"
               :: "v"(lds_addr), "v"(ga) : "memory");
}

__device__ __forceinline__ v16bf loadA(const unsigned short* p) {
  v8bf alo = *(const v8bf*)(p);
  v8bf ahi = *(const v8bf*)(p + 8);
  return __builtin_shufflevector(alo, ahi,
           0, 1, 2, 3, 4, 5, 6, 7, 8, 9, 10, 11, 12, 13, 14, 15);
}

// ----------------------- persistent LSTM kernel ----------------------------
__global__ void __launch_bounds__(NTHREADS_)
lstm_persistent(unsigned char* __restrict__ ws,
                const float* __restrict__ bias,
                float* __restrict__ out) {
  __shared__ __align__(16) unsigned short smem[SMEM_X_ELEMS + SMEM_H_ELEMS];

  unsigned* bar = (unsigned*)(ws + OFF_BAR);
  const unsigned short* wfrag  = (const unsigned short*)(ws + OFF_WFRAG);
  const unsigned short* featsb = (const unsigned short*)(ws + OFF_FEATSB);
  unsigned short*       hb     = (unsigned short*)(ws + OFF_HB);
  float*                cst    = (float*)(ws + OFF_C);
  float*                gates  = (float*)(ws + OFF_GATES);

  const int tid  = (int)threadIdx.x;
  const int lane = tid & 31;
  const int wave = tid >> 5;
  const int wid  = (int)blockIdx.x * (NTHREADS_ / 32) + wave;  // 0..511
  const int nt   = wid & (NT_ - 1);        // N-tile (gate column / 16)
  const int mtB  = (int)blockIdx.x >> 5;   // M-tile, uniform per block
  const int sel  = lane >> 4;              // lane group (A/B K-halves)
  const int mloc = lane & 15;

  // ---- load this wave's 48 B fragments into registers (384 VGPRs) ----
  const unsigned short* wf = wfrag + (size_t)nt * 512 + (size_t)lane * 16;
  v16bf breg[KB_];
#pragma unroll
  for (int kb = 0; kb < KB_; ++kb)
    breg[kb] = *(const v16bf*)(wf + (size_t)kb * (NT_ * 512));

  // ---- precompute per-thread staging addresses (hoisted out of t-loop) ----
  // x_t: 16 rows x 512 bf16 = 1024 16B-chunks; h: 16 rows x 1024 = 2048
  unsigned xoff[4];
  const unsigned short* xsrc[4];
#pragma unroll
  for (int i = 0; i < 4; ++i) {
    int chunk = tid + i * NTHREADS_;
    int row = chunk >> 6, c8 = chunk & 63;
    xoff[i] = (unsigned)(unsigned long long)(smem + row * XPITCH + c8 * 8);
    xsrc[i] = featsb + ((size_t)(mtB * 16 + row) * T_) * D_ + c8 * 8;  // t=0
  }
  unsigned hoff[8];
  const unsigned short* hsrc[8];
#pragma unroll
  for (int i = 0; i < 8; ++i) {
    int chunk = tid + i * NTHREADS_;
    int row = chunk >> 7, c8 = chunk & 127;
    hoff[i] = (unsigned)(unsigned long long)(smem + SMEM_X_ELEMS +
                                             row * HPITCH + c8 * 8);
    hsrc[i] = hb + (size_t)(mtB * 16 + row) * H_ + c8 * 8;
  }

  // per-lane A fragment bases in LDS
  const unsigned short* axl = smem + (size_t)mloc * XPITCH + 16 * sel;
  const unsigned short* ahl = smem + SMEM_X_ELEMS + (size_t)mloc * HPITCH
                            + 16 * sel;
  const unsigned gid = blockIdx.x * NTHREADS_ + threadIdx.x;  // 0..16383

  for (int t = 0; t < T_; ++t) {
    // ---- stage this block's A rows into LDS (async global->LDS) ----
#pragma unroll
    for (int i = 0; i < 4; ++i) {
      asyncCopy16(xoff[i], xsrc[i]);
      xsrc[i] += D_;  // next timestep's x row
    }
#pragma unroll
    for (int i = 0; i < 8; ++i) asyncCopy16(hoff[i], hsrc[i]);
    asm volatile("s_wait_asynccnt 0x0" ::: "memory");
    __syncthreads();

    // ---- phase 1: gates[32 x 4096] = [x_t || h_prev] * [Wx; Wh] ----
    v8f acc = {};
#pragma unroll
    for (int kb = 0; kb < KBX_; ++kb) {
      v16bf a = loadA(axl + kb * 32);
      acc = __builtin_amdgcn_wmma_f32_16x16x32_bf16(
          false, a, false, breg[kb], (short)0, acc, false, false);
    }
#pragma unroll
    for (int kb = 0; kb < KBH_; ++kb) {
      v16bf a = loadA(ahl + kb * 32);
      acc = __builtin_amdgcn_wmma_f32_16x16x32_bf16(
          false, a, false, breg[KBX_ + kb], (short)0, acc, false, false);
    }
    // C/D layout: VGPR r, lanes 0-15 -> M=r, lanes 16-31 -> M=8+r; N=lane&15
    {
      const int n = nt * 16 + mloc;
      const int mrow = mtB * 16 + 8 * sel;
      float* gp = gates + n;
#pragma unroll
      for (int r = 0; r < 8; ++r) gp[(size_t)(mrow + r) * G_] = acc[r];
    }

    gridBarrier(bar);

    // ---- phase 2: gate nonlinearity + state update (2 elems / thread) ----
#pragma unroll
    for (int p = 0; p < 2; ++p) {
      const unsigned idx = gid + (unsigned)p * (NBLOCKS_ * NTHREADS_);
      const int mm = (int)(idx >> 10);
      const int j  = (int)(idx & (H_ - 1));
      const float* gr = gates + (size_t)mm * G_;
      float ig = gr[j]            + bias[j];
      float fg = gr[H_ + j]       + bias[H_ + j];
      float gg = gr[2 * H_ + j]   + bias[2 * H_ + j];
      float og = gr[3 * H_ + j]   + bias[3 * H_ + j];
      float it = 1.0f / (1.0f + __expf(-ig));
      float ft = 1.0f / (1.0f + __expf(-fg));
      float gt = tanhf(gg);
      float ot = 1.0f / (1.0f + __expf(-og));
      float cn = tanhf(ft * cst[idx] + it * gt);   // reference quirk
      cst[idx] = cn;
      float hn = ot * cn;
      hb[idx] = f2bfbits(hn);                       // next step's A operand
      out[((size_t)mm * T_ + t) * H_ + j] = hn;     // [B, T, H]
    }

    gridBarrier(bar);
  }
}

// --------------------------- launcher --------------------------------------
extern "C" void kernel_launch(void* const* d_in, const int* in_sizes, int n_in,
                              void* d_out, int out_size, void* d_ws,
                              size_t ws_size, hipStream_t stream) {
  (void)in_sizes; (void)n_in; (void)out_size; (void)ws_size;
  const float* feats = (const float*)d_in[0];   // [32, 512, 512]
  const float* Wx    = (const float*)d_in[1];   // [512, 4096]
  const float* Wh    = (const float*)d_in[2];   // [1024, 4096]
  const float* bias  = (const float*)d_in[3];   // [4096]
  float* out = (float*)d_out;                   // [32, 512, 1024]
  unsigned char* ws = (unsigned char*)d_ws;

  const unsigned prepThreads = 256;
  const unsigned prepBlocks = (unsigned)((FEATSB_ELEMS + prepThreads - 1) /
                                         prepThreads);
  lstm_prep<<<prepBlocks, prepThreads, 0, stream>>>(feats, Wx, Wh, ws);
  lstm_persistent<<<NBLOCKS_, NTHREADS_, 0, stream>>>(ws, bias, out);
}
